// Jordan_74440373174741
// MI455X (gfx1250) — compile-verified, matrix operand order checked by reading, not run on hardware
//
#include <hip/hip_runtime.h>
#include <hip/hip_bf16.h>
#include <math.h>

// ---------------------------------------------------------------------------
// Jordan RNN on MI455X (gfx1250, wave32, WMMA).
//   Phase P (prep):   pack Whx/Wyh/Wout fp32 -> bf16 in WMMA B-register layout,
//                     convert y0 = last_logits to bf16, reset barrier.
//   Phase A (gemm):   xh[b,t,:] = emb @ Whx^T + bhx   -> stored into d_out logits.
//   Phase S (scan):   persistent 16-WG cooperative kernel, 2048 steps; B tiles
//                     cached in LDS, A fragments fully preloaded per GEMM.
// ---------------------------------------------------------------------------

typedef __attribute__((ext_vector_type(16))) __bf16 v16bf;
typedef __attribute__((ext_vector_type(8)))  __bf16 v8bf;
typedef __attribute__((ext_vector_type(8)))  float  v8f;
typedef __attribute__((ext_vector_type(4)))  float  v4f;

#define BB 64
#define TT 2048
#define KK 512      // D == H == O == 512
#define NWG_SCAN 16

static __device__ __forceinline__ v8f vzero8() {
  v8f z;
#pragma unroll
  for (int i = 0; i < 8; ++i) z[i] = 0.0f;
  return z;
}

static __device__ __forceinline__ v8f wmma_bf16(v16bf a, v16bf b, v8f c) {
  // v_wmma_f32_16x16x32_bf16  D = A*B + C
  return __builtin_amdgcn_wmma_f32_16x16x32_bf16(false, a, false, b, (short)0, c,
                                                 false, false);
}

static __device__ __forceinline__ v16bf cat8(v8bf lo, v8bf hi) {
  return __builtin_shufflevector(lo, hi, 0, 1, 2, 3, 4, 5, 6, 7,
                                 8, 9, 10, 11, 12, 13, 14, 15);
}

// ---------------------------------------------------------------------------
// Prep: pack 3 weight matrices into WMMA-B layout + bf16 y0 + barrier reset.
// Packed element r = ((kt*32 + nt)*32 + lane)*16 + j holds
//   W[n*512 + k], n = nt*16 + (lane&15), k = kt*32 + (j&7) + 8*(lane>>4) + 16*(j>>3)
// (16-bit operand register layout, ISA 7.12.2).
// ---------------------------------------------------------------------------
__global__ __launch_bounds__(256) void jordan_prep(
    const float* __restrict__ Whx, const float* __restrict__ Wyh,
    const float* __restrict__ Wout, const float* __restrict__ last_logits,
    __bf16* __restrict__ WhxB, __bf16* __restrict__ WyhB,
    __bf16* __restrict__ WoutB, __bf16* __restrict__ yb,
    unsigned* __restrict__ bar) {
  const size_t NPACK = (size_t)KK * KK;  // 262144
  size_t idx = (size_t)blockIdx.x * 256 + threadIdx.x;
  if (idx < 3 * NPACK) {
    int m = (int)(idx / NPACK);
    size_t r = idx % NPACK;
    int j = (int)(r & 15);
    int lane = (int)((r >> 4) & 31);
    int ktnt = (int)(r >> 9);
    int nt = ktnt & 31;
    int kt = ktnt >> 5;
    int half = lane >> 4;
    int k = kt * 32 + (j & 7) + 8 * half + 16 * (j >> 3);
    int n = nt * 16 + (lane & 15);
    const float* W = (m == 0) ? Whx : (m == 1) ? Wyh : Wout;
    __bf16* Dst = (m == 0) ? WhxB : (m == 1) ? WyhB : WoutB;
    Dst[r] = (__bf16)W[(size_t)n * KK + k];
  } else if (idx < 3 * NPACK + (size_t)BB * KK) {
    size_t r = idx - 3 * NPACK;
    yb[r] = (__bf16)last_logits[r];
  } else if (idx == 3 * NPACK + (size_t)BB * KK) {
    bar[0] = 0u;
    bar[1] = 0u;
  }
}

// ---------------------------------------------------------------------------
// Phase A: xh = emb @ Whx^T + bhx.  Grid: 2048 WGs x 256 thr (8 waves).
// WG tile = 64 rows x 512 cols (embeddings rows read exactly once per WG).
// Wave w: row-pair mp = w&1 (2 M-tiles), N-quad nq = w>>1 (8 N-tiles).
// Each B tile feeds 2 WMMAs; 8 B loads + 8 A loads per kt vs 16 WMMAs.
// ---------------------------------------------------------------------------
__global__ __launch_bounds__(256) void jordan_phaseA(
    const float* __restrict__ emb, const float* __restrict__ bhx,
    const __bf16* __restrict__ WhxB, float* __restrict__ out) {
  const int lane = threadIdx.x & 31;
  const int wave = threadIdx.x >> 5;
  const int mp = wave & 1;        // which pair of M-tiles
  const int nq = wave >> 1;       // 0..3 -> 8 N-tiles each
  const int half = lane >> 4;
  const int rowBase = blockIdx.x * 64 + mp * 32;   // flat (b*T + t)
  const int r0 = rowBase + (lane & 15);

  v8f acc[2][8];
#pragma unroll
  for (int mi = 0; mi < 2; ++mi)
#pragma unroll
    for (int nj = 0; nj < 8; ++nj) acc[mi][nj] = vzero8();

  for (int kt = 0; kt < 16; ++kt) {
    v16bf a[2];
#pragma unroll
    for (int mi = 0; mi < 2; ++mi) {
      const float* ap = emb + (size_t)(r0 + mi * 16) * KK + kt * 32 + 8 * half;
      v4f f0 = *(const v4f*)(ap + 0);
      v4f f1 = *(const v4f*)(ap + 4);
      v4f f2 = *(const v4f*)(ap + 16);
      v4f f3 = *(const v4f*)(ap + 20);
#pragma unroll
      for (int i = 0; i < 4; ++i) {
        a[mi][i + 0]  = (__bf16)f0[i];
        a[mi][i + 4]  = (__bf16)f1[i];
        a[mi][i + 8]  = (__bf16)f2[i];
        a[mi][i + 12] = (__bf16)f3[i];
      }
    }
    v16bf bb[8];
#pragma unroll
    for (int nj = 0; nj < 8; ++nj)
      bb[nj] = *(const v16bf*)(WhxB +
               (((size_t)(kt * 32 + nq * 8 + nj)) * 32 + lane) * 16);
#pragma unroll
    for (int nj = 0; nj < 8; ++nj)
#pragma unroll
      for (int mi = 0; mi < 2; ++mi)
        acc[mi][nj] = wmma_bf16(a[mi], bb[nj], acc[mi][nj]);
  }

#pragma unroll
  for (int nj = 0; nj < 8; ++nj) {
    int n = (nq * 8 + nj) * 16 + (lane & 15);
    float bias = bhx[n];
#pragma unroll
    for (int mi = 0; mi < 2; ++mi)
#pragma unroll
      for (int r = 0; r < 8; ++r) {
        int row = rowBase + mi * 16 + r + 8 * half;  // C layout: M = r + 8*half
        out[(size_t)row * KK + n] = acc[mi][nj][r] + bias;
      }
  }
}

// ---------------------------------------------------------------------------
// Grid-wide barrier (persistent scan kernel, NWG_SCAN workgroups).
// ---------------------------------------------------------------------------
static __device__ __forceinline__ void grid_barrier(unsigned* bar) {
  __builtin_amdgcn_fence(__ATOMIC_RELEASE, "agent");
  __syncthreads();
  if (threadIdx.x == 0) {
    unsigned gen = __hip_atomic_load(&bar[1], __ATOMIC_RELAXED,
                                     __HIP_MEMORY_SCOPE_AGENT);
    unsigned prev = __hip_atomic_fetch_add(&bar[0], 1u, __ATOMIC_ACQ_REL,
                                           __HIP_MEMORY_SCOPE_AGENT);
    if (prev == NWG_SCAN - 1) {
      __hip_atomic_store(&bar[0], 0u, __ATOMIC_RELAXED,
                         __HIP_MEMORY_SCOPE_AGENT);
      __hip_atomic_fetch_add(&bar[1], 1u, __ATOMIC_RELEASE,
                             __HIP_MEMORY_SCOPE_AGENT);
    } else {
      while (__hip_atomic_load(&bar[1], __ATOMIC_ACQUIRE,
                               __HIP_MEMORY_SCOPE_AGENT) == gen) {
        __builtin_amdgcn_s_sleep(1);
      }
    }
  }
  __syncthreads();
  __builtin_amdgcn_fence(__ATOMIC_ACQUIRE, "agent");
}

// One 16x16 output tile, K=512: all 16 A fragments preloaded (32 b128 loads
// in flight), B tiles served from LDS (immune to the barrier's L0 invalidate).
static __device__ __forceinline__ v8f gemm_lds(const __bf16* __restrict__ act,
                                               const v16bf Bt[16][32],
                                               int arow, int half, int lane) {
  v8bf a0[16], a1[16];
#pragma unroll
  for (int kt = 0; kt < 16; ++kt) {
    const __bf16* ap = act + (size_t)arow * KK + kt * 32 + 8 * half;
    a0[kt] = *(const v8bf*)(ap);
    a1[kt] = *(const v8bf*)(ap + 16);
  }
  v8f acc = vzero8();
#pragma unroll
  for (int kt = 0; kt < 16; ++kt)
    acc = wmma_bf16(cat8(a0[kt], a1[kt]), Bt[kt][lane], acc);
  return acc;
}

// ---------------------------------------------------------------------------
// Phase S: persistent scan. 16 WGs x 256 thr. WG owns 32 cols (2 N-tiles);
// wave w: mtile = w&3 (batch rows), ntile = w>>2. Wyh/Wout slices in LDS.
// ---------------------------------------------------------------------------
__global__ __launch_bounds__(256) void jordan_scan(
    const float* __restrict__ byh, const float* __restrict__ bout,
    const __bf16* __restrict__ WyhB, const __bf16* __restrict__ WoutB,
    __bf16* __restrict__ yb, __bf16* __restrict__ hb,
    unsigned* __restrict__ bar, float* __restrict__ out,
    float* __restrict__ last) {
  __shared__ v16bf Bsh[2][2][16][32];  // [gemm][ntile][kt][lane] = 64 KiB

  const int lane = threadIdx.x & 31;
  const int wave = threadIdx.x >> 5;
  const int mtile = wave & 3;
  const int ntile = wave >> 2;               // 0..1
  const int half = lane >> 4;
  const int nt = blockIdx.x * 2 + ntile;     // global N tile 0..31
  const int n = nt * 16 + (lane & 15);
  const int arow = mtile * 16 + (lane & 15);
  const float byh_n = byh[n];
  const float bout_n = bout[n];

  // ---- Cooperative fill of LDS with this WG's Wyh/Wout B-tile slices -------
#pragma unroll
  for (int i = 0; i < 16; ++i) {
    int cc = threadIdx.x + i * 256;          // 16-byte chunk id, 0..4095
    int ti = cc >> 6;                        // tile 0..63
    int w = cc & 63;                         // chunk within 1 KiB tile
    int g = ti >> 5;                         // 0 = Wyh, 1 = Wout
    int nti = (ti >> 4) & 1;
    int kt = ti & 15;
    int ln = w >> 1;
    int h16 = w & 1;
    const __bf16* srcm = g ? WoutB : WyhB;
    const __bf16* src = srcm +
        (((size_t)(kt * 32 + blockIdx.x * 2 + nti)) * 32 + ln) * 16 + h16 * 8;
    v8bf v = *(const v8bf*)src;
    *((v8bf*)&Bsh[g][nti][kt][ln] + h16) = v;
  }
  __syncthreads();

  for (int t = 0; t < TT; ++t) {
    // Prefetch xh_t early (only this wave ever touches these addresses).
    float xv[8];
#pragma unroll
    for (int r = 0; r < 8; ++r) {
      int row = mtile * 16 + r + 8 * half;
      xv[r] = out[((size_t)row * TT + t) * KK + n];
    }

    // ---- GEMM1: h = tanh(xh_t + y @ Wyh^T + byh) --------------------------
    v8f acc = gemm_lds(yb, Bsh[0][ntile], arow, half, lane);
#pragma unroll
    for (int r = 0; r < 8; ++r) {
      int row = mtile * 16 + r + 8 * half;   // batch index b
      float hv = tanhf(acc[r] + xv[r] + byh_n);
      hb[(size_t)row * KK + n] = (__bf16)hv;
    }
    grid_barrier(bar);

    // ---- GEMM2: y = h @ Wout^T + bout -------------------------------------
    acc = gemm_lds(hb, Bsh[1][ntile], arow, half, lane);
#pragma unroll
    for (int r = 0; r < 8; ++r) {
      int row = mtile * 16 + r + 8 * half;
      float yv = acc[r] + bout_n;
      out[((size_t)row * TT + t) * KK + n] = yv;   // logits (overwrites xh)
      yb[(size_t)row * KK + n] = (__bf16)yv;       // feeds next step
      if (t == TT - 1) last[(size_t)row * KK + n] = yv;
    }
    grid_barrier(bar);
  }
}

// ---------------------------------------------------------------------------
extern "C" void kernel_launch(void* const* d_in, const int* in_sizes, int n_in,
                              void* d_out, int out_size, void* d_ws,
                              size_t ws_size, hipStream_t stream) {
  const float* emb  = (const float*)d_in[0];  // [B,T,D]
  const float* y0   = (const float*)d_in[1];  // [B,O]
  const float* Whx  = (const float*)d_in[2];  // [H,D]
  const float* bhx  = (const float*)d_in[3];  // [H]
  const float* Wyh  = (const float*)d_in[4];  // [H,O]
  const float* byh  = (const float*)d_in[5];  // [H]
  const float* Wout = (const float*)d_in[6];  // [O,H]
  const float* bout = (const float*)d_in[7];  // [O]
  float* out = (float*)d_out;                             // logits [B,T,O]
  float* last = out + (size_t)BB * TT * KK;               // last   [B,O]

  uint8_t* ws = (uint8_t*)d_ws;
  unsigned* bar  = (unsigned*)ws;                          //   256 B
  __bf16*  yb    = (__bf16*)(ws + 256);                    //  64 KiB
  __bf16*  hb    = (__bf16*)(ws + 256 + 65536);            //  64 KiB
  __bf16*  WhxB  = (__bf16*)(ws + 256 + 2 * 65536);        // 512 KiB each
  __bf16*  WyhB  = WhxB + (size_t)KK * KK;
  __bf16*  WoutB = WyhB + (size_t)KK * KK;

  // Prep: 3*262144 pack elems + 32768 y0 elems + 1 barrier slot.
  {
    size_t total = 3 * (size_t)KK * KK + (size_t)BB * KK + 1;
    int blocks = (int)((total + 255) / 256);
    jordan_prep<<<blocks, 256, 0, stream>>>(Whx, Wyh, Wout, y0, WhxB, WyhB,
                                            WoutB, yb, bar);
  }
  // Phase A: 131072 rows / 64 per WG = 2048 WGs.
  jordan_phaseA<<<(BB * TT) / 64, 256, 0, stream>>>(emb, bhx, WhxB, out);
  // Phase S: persistent cooperative scan.
  jordan_scan<<<NWG_SCAN, 256, 0, stream>>>(byh, bout, WyhB, WoutB, yb, hb,
                                            bar, out, last);
}
